// DecoderIGCN_41558103556711
// MI455X (gfx1250) — matrix operand order repaired and checked by reading, not compiled, not run on hardware
//
#include <hip/hip_runtime.h>
#include <hip/hip_bf16.h>

typedef __attribute__((ext_vector_type(16))) __bf16 v16bf;
typedef __attribute__((ext_vector_type(8)))  __bf16 v8bf;
typedef __attribute__((ext_vector_type(8)))  float  v8f;
typedef int v4i_ __attribute__((vector_size(16)));   // b128 payload type for async-LDS builtin

#define NE   512
#define DD   768
#define RD   100
#define RH   8
#define NL   2
#define DR   96      // D / R
#define CP1  128     // conv1 input channels padded (100 -> 128)
#define OC1  200
#define OC1P 208     // conv1 out channels padded to mult of 16
#define CP2  224     // conv2 input channels padded (200 -> 224, mult of 32)
#define OC2  100
#define OC2P 112
#define NP   514     // spatial with 1-px zero halo
#define EPSL 1e-5f

#ifndef __has_builtin
#define __has_builtin(x) 0
#endif
#if __has_builtin(__builtin_amdgcn_global_load_async_to_lds_b128) && \
    __has_builtin(__builtin_amdgcn_s_wait_asynccnt)
#define ASYNC_LDS 1
#else
#define ASYNC_LDS 0
#endif

// ---------------- WMMA fragment helpers (gfx1250 wave32 layouts) ----------------

__device__ __forceinline__ v8f vzero8() {
  v8f v;
#pragma unroll
  for (int i = 0; i < 8; ++i) v[i] = 0.0f;
  return v;
}

// A: 16x32 bf16, row-major source. Lane handles row (lane&15).
// lo lanes: K = kb+0..7 then kb+16..23 ; hi lanes: K = kb+8..15 then kb+24..31.
__device__ __forceinline__ v16bf frag_a(const __bf16* rowp, int kb, int hi) {
  v8bf c0 = *(const v8bf*)(rowp + kb + hi * 8);
  v8bf c1 = *(const v8bf*)(rowp + kb + 16 + hi * 8);
  v16bf r;
#pragma unroll
  for (int j = 0; j < 8; ++j) { r[j] = c0[j]; r[j + 8] = c1[j]; }
  return r;
}

// B: 32x16 bf16, stored NT (row-major [n][k]). Lane handles col n=(lane&15);
// lo lanes: K = kb+0..7 then kb+16..23 ; hi lanes: K = kb+8..15 then kb+24..31.
__device__ __forceinline__ v16bf frag_b(const __bf16* rowp, int kb, int hi) {
  v8bf c0 = *(const v8bf*)(rowp + kb + hi * 8);
  v8bf c1 = *(const v8bf*)(rowp + kb + 16 + hi * 8);
  v16bf r;
#pragma unroll
  for (int j = 0; j < 8; ++j) { r[j] = c0[j]; r[j + 8] = c1[j]; }
  return r;
}

__device__ __forceinline__ v8f wmma_bf(v16bf a, v16bf b, v8f c) {
  return __builtin_amdgcn_wmma_f32_16x16x32_bf16(false, a, false, b, (short)0, c,
                                                 false, false);
}
// C/D map: element g of lane L -> (m = (L>>4)*8+g, n = L&15) within 16x16 tile.

// ---------------- elementwise / transpose utility kernels ----------------

__global__ void k_cvt_bf16(const float* __restrict__ in, __bf16* __restrict__ out,
                           size_t n) {
  size_t i = (size_t)blockIdx.x * blockDim.x + threadIdx.x;
  size_t st = (size_t)gridDim.x * blockDim.x;
  for (; i < n; i += st) out[i] = (__bf16)in[i];
}

__global__ void k_fill0_bf16(__bf16* __restrict__ out, size_t n) {
  size_t i = (size_t)blockIdx.x * blockDim.x + threadIdx.x;
  size_t st = (size_t)gridDim.x * blockDim.x;
  for (; i < n; i += st) out[i] = (__bf16)0.0f;
}

// in fp32 [z][rows][cols] -> out bf16 [z][cols][rows]
__global__ void k_transpose_cvt(const float* __restrict__ in, __bf16* __restrict__ out,
                                int rows, int cols) {
  __shared__ float tile[32][33];
  const size_t plane = (size_t)rows * cols;
  in  += plane * blockIdx.z;
  out += plane * blockIdx.z;
  int c0 = blockIdx.x * 32, r0 = blockIdx.y * 32;
  int tx = threadIdx.x, ty = threadIdx.y;  // 32 x 8
  for (int i = ty; i < 32; i += 8) {
    int r = r0 + i, c = c0 + tx;
    tile[i][tx] = (r < rows && c < cols) ? in[(size_t)r * cols + c] : 0.0f;
  }
  __syncthreads();
  for (int i = ty; i < 32; i += 8) {
    int c = c0 + i, r = r0 + tx;
    if (r < rows && c < cols) out[(size_t)c * rows + r] = (__bf16)tile[tx][i];
  }
}

// ---------------- cooperative LDS double-buffered GEMM (NT, bf16 WMMA) ----------------
// Block = 128 threads (4 waves), 64x64 output tile, k-step 32.
// MODE 0: C bf16 [m*ldc+n] (+z*cStr)           (T = E x Rt^T)
// MODE 1: rel fp32 scatter [((m*NE+n)*RD)+rbase+z], relu   (rel = T x E^T)
// MODE 2: C fp32 [m*ldc+n] + bias[n]           (ent MLP)

#define TM 64
#define TN 64
#define TK 32

__device__ __forceinline__ void tile_copy(const __bf16* __restrict__ gA,
                                          const __bf16* __restrict__ gB,
                                          int lda, int ldb, int kc,
                                          __bf16* sA, __bf16* sB, int tid) {
#pragma unroll
  for (int h = 0; h < 2; ++h) {
    int c = tid + h * 128;          // 0..255 -> 64 rows x 4 16B-chunks
    int row = c >> 2, off = (c & 3) * 8;
    const __bf16* gsa = gA + (size_t)row * lda + kc * TK + off;
    const __bf16* gsb = gB + (size_t)row * ldb + kc * TK + off;
    __bf16* lsa = sA + row * TK + off;
    __bf16* lsb = sB + row * TK + off;
#if ASYNC_LDS
    __builtin_amdgcn_global_load_async_to_lds_b128(
        (v4i_*)gsa, (__attribute__((address_space(3))) v4i_*)lsa, 0, 0);
    __builtin_amdgcn_global_load_async_to_lds_b128(
        (v4i_*)gsb, (__attribute__((address_space(3))) v4i_*)lsb, 0, 0);
#else
    *(v8bf*)lsa = *(const v8bf*)gsa;
    *(v8bf*)lsb = *(const v8bf*)gsb;
#endif
  }
}

__device__ __forceinline__ void wait_async_copies() {
#if ASYNC_LDS
  __builtin_amdgcn_s_wait_asynccnt(0);
#endif
}

template <int MODE>
__global__ void k_gemm_coop(const __bf16* __restrict__ A, const __bf16* __restrict__ B,
                            void* __restrict__ Cv, const float* __restrict__ bias,
                            int K, int lda, int ldb, int ldc,
                            long aStr, long bStr, long cStr, int rbase) {
  __shared__ __bf16 As[2][TM * TK];
  __shared__ __bf16 Bs[2][TN * TK];
  const int tid = threadIdx.x;
  const int lane = tid & 31, hi = lane >> 4, lm = lane & 15;
  const int wave = tid >> 5;
  const int wm = (wave >> 1) * 32, wn = (wave & 1) * 32;
  const int z = blockIdx.z;
  const int mBase = blockIdx.y * TM, nBase = blockIdx.x * TN;
  const __bf16* gA = A + (size_t)z * aStr + (size_t)mBase * lda;
  const __bf16* gB = B + (size_t)z * bStr + (size_t)nBase * ldb;

  v8f acc[2][2];
#pragma unroll
  for (int i = 0; i < 2; ++i)
#pragma unroll
    for (int j = 0; j < 2; ++j) acc[i][j] = vzero8();

  const int nk = K / TK;
  tile_copy(gA, gB, lda, ldb, 0, As[0], Bs[0], tid);
  for (int kc = 0; kc < nk; ++kc) {
    wait_async_copies();
    __syncthreads();
    if (kc + 1 < nk)
      tile_copy(gA, gB, lda, ldb, kc + 1, As[(kc + 1) & 1], Bs[(kc + 1) & 1], tid);
    const __bf16* sa = As[kc & 1];
    const __bf16* sb = Bs[kc & 1];
    v16bf a0 = frag_a(sa + (wm + lm) * TK, 0, hi);
    v16bf a1 = frag_a(sa + (wm + 16 + lm) * TK, 0, hi);
    v16bf b0 = frag_b(sb + (wn + lm) * TK, 0, hi);
    v16bf b1 = frag_b(sb + (wn + 16 + lm) * TK, 0, hi);
    acc[0][0] = wmma_bf(a0, b0, acc[0][0]);
    acc[0][1] = wmma_bf(a0, b1, acc[0][1]);
    acc[1][0] = wmma_bf(a1, b0, acc[1][0]);
    acc[1][1] = wmma_bf(a1, b1, acc[1][1]);
  }

#pragma unroll
  for (int mi = 0; mi < 2; ++mi)
#pragma unroll
    for (int ni = 0; ni < 2; ++ni)
#pragma unroll
      for (int g = 0; g < 8; ++g) {
        const int m = mBase + wm + mi * 16 + hi * 8 + g;
        const int n = nBase + wn + ni * 16 + lm;
        const float v = acc[mi][ni][g];
        if (MODE == 0) {
          ((__bf16*)Cv + (size_t)z * cStr)[(size_t)m * ldc + n] = (__bf16)v;
        } else if (MODE == 1) {
          ((float*)Cv)[((size_t)m * NE + n) * RD + rbase + z] = fmaxf(v, 0.0f);
        } else {
          ((float*)Cv)[(size_t)m * ldc + n] = v + bias[n];
        }
      }
}

// ---------------- register-resident pipelined NT GEMM core ----------------

template <int NT>
__device__ __forceinline__ void reg_gemm_nt(const __bf16* __restrict__ arow,
                                            const __bf16* __restrict__ brow,
                                            int ldb, int K, int hi, v8f acc[NT]) {
  v16bf a = frag_a(arow, 0, hi);
  v16bf b[NT];
#pragma unroll
  for (int t = 0; t < NT; ++t) b[t] = frag_b(brow + (size_t)t * 16 * ldb, 0, hi);
  for (int kb = 0; kb < K; kb += 32) {
    v16bf an = a;
    v16bf bn[NT];
#pragma unroll
    for (int t = 0; t < NT; ++t) bn[t] = b[t];
    if (kb + 32 < K) {
      an = frag_a(arow, kb + 32, hi);
#pragma unroll
      for (int t = 0; t < NT; ++t)
        bn[t] = frag_b(brow + (size_t)t * 16 * ldb, kb + 32, hi);
    }
#pragma unroll
    for (int t = 0; t < NT; ++t) acc[t] = wmma_bf(a, b[t], acc[t]);
    a = an;
#pragma unroll
    for (int t = 0; t < NT; ++t) b[t] = bn[t];
  }
}

// ---------------- LayerNorm over RDIM per (i,j); optional residual add ----------------

__global__ void k_add_ln_rdim(float* __restrict__ dst, const float* __restrict__ x,
                              const float* __restrict__ res,
                              const float* __restrict__ g, const float* __restrict__ b) {
  const int wave = threadIdx.x >> 5, lane = threadIdx.x & 31;
  const size_t pix = (size_t)blockIdx.x * (blockDim.x >> 5) + wave;
  if (pix >= (size_t)NE * NE) return;
  const float* xp = x + pix * RD;
  const float* rp = res ? res + pix * RD : nullptr;
  float v[4];
  float s = 0.0f, s2 = 0.0f;
#pragma unroll
  for (int k = 0; k < 4; ++k) {
    int c = lane + 32 * k;
    float t = 0.0f;
    if (c < RD) { t = xp[c]; if (rp) t += rp[c]; }
    v[k] = t; s += t; s2 += t * t;
  }
#pragma unroll
  for (int m = 16; m; m >>= 1) { s += __shfl_xor(s, m); s2 += __shfl_xor(s2, m); }
  const float mean = s / (float)RD;
  const float var = s2 / (float)RD - mean * mean;
  const float rs = rsqrtf(var + EPSL);
  float* dp = dst + pix * RD;
#pragma unroll
  for (int k = 0; k < 4; ++k) {
    int c = lane + 32 * k;
    if (c < RD) dp[c] = (v[k] - mean) * rs * g[c] + b[c];
  }
}

// ---------------- conv FFN: packing + implicit-GEMM conv ----------------

// w fp32 [OC][IC][3][3] -> wp bf16 [OCp][tap=9][Cp] (zero padded)
__global__ void k_pack_w(const float* __restrict__ w, __bf16* __restrict__ wp,
                         int OCp, int Cp, int OC, int IC) {
  size_t n = (size_t)OCp * 9 * Cp;
  size_t i = (size_t)blockIdx.x * blockDim.x + threadIdx.x;
  size_t st = (size_t)gridDim.x * blockDim.x;
  for (; i < n; i += st) {
    int ic = (int)(i % Cp);
    int tap = (int)((i / Cp) % 9);
    int oc = (int)(i / ((size_t)Cp * 9));
    float v = 0.0f;
    if (oc < OC && ic < IC) v = w[((size_t)oc * IC + ic) * 9 + tap];
    wp[i] = (__bf16)v;
  }
}

// rel fp32 [512][512][RD] -> P1 bf16 [NP][NP][CP1] with zero halo + channel pad
__global__ void k_pack_in(const float* __restrict__ src, __bf16* __restrict__ dst) {
  size_t n = (size_t)NP * NP * CP1;
  size_t i = (size_t)blockIdx.x * blockDim.x + threadIdx.x;
  size_t st = (size_t)gridDim.x * blockDim.x;
  for (; i < n; i += st) {
    int c = (int)(i % CP1);
    size_t q = i / CP1;
    int x = (int)(q % NP);
    int y = (int)(q / NP);
    float v = 0.0f;
    if (c < RD && y >= 1 && y <= NE && x >= 1 && x <= NE)
      v = src[((size_t)(y - 1) * NE + (x - 1)) * RD + c];
    dst[i] = (__bf16)v;
  }
}

// software-pipelined per-tap implicit GEMM accumulation
template <int CP>
__device__ __forceinline__ void conv_acc(const __bf16* __restrict__ W9,
                                         const __bf16* __restrict__ P,
                                         int y, int x, int hi, v8f& acc) {
  constexpr int KS = CP / 32;
  constexpr int TOT = 9 * KS;
  v16bf a = frag_a(W9, 0, hi);
  v16bf b = frag_b(P + ((size_t)y * NP + x) * CP, 0, hi);
#pragma unroll
  for (int it = 0; it < TOT; ++it) {
    v16bf an = a, bn = b;
    if (it + 1 < TOT) {
      const int nxt = it + 1;
      const int tap = nxt / KS, kb = (nxt % KS) * 32;
      const int ky = tap / 3, kx = tap - ky * 3;
      an = frag_a(W9 + tap * CP, kb, hi);
      bn = frag_b(P + ((size_t)(y + ky) * NP + (x + kx)) * CP, kb, hi);
    }
    acc = wmma_bf(a, b, acc);
    a = an; b = bn;
  }
}

// conv1: P1 [NP][NP][CP1] -> P2 [NP][NP][CP2] (relu+bias, bf16, interior only)
__global__ void k_conv1(const __bf16* __restrict__ Wp, const __bf16* __restrict__ P1,
                        const float* __restrict__ bias, __bf16* __restrict__ P2) {
  const int wave = threadIdx.x >> 5, lane = threadIdx.x & 31;
  const int hi = lane >> 4, lm = lane & 15;
  const int n = (blockIdx.x * 4 + wave) * 16 + lm;
  const int y = n >> 9, x = n & 511;
  const int mb = blockIdx.y * 16;
  const __bf16* arow = Wp + (size_t)(mb + lm) * (9 * CP1);
  v8f acc = vzero8();
  conv_acc<CP1>(arow, P1, y, x, hi, acc);
#pragma unroll
  for (int g = 0; g < 8; ++g) {
    int m = mb + hi * 8 + g;
    if (m < OC1) {
      float v = fmaxf(acc[g] + bias[m], 0.0f);
      P2[((size_t)(y + 1) * NP + (x + 1)) * CP2 + m] = (__bf16)v;
    }
  }
}

// conv2: P2 -> c2out fp32 [512][512][RD] (+bias, no relu)
__global__ void k_conv2(const __bf16* __restrict__ Wp, const __bf16* __restrict__ P2,
                        const float* __restrict__ bias, float* __restrict__ out) {
  const int wave = threadIdx.x >> 5, lane = threadIdx.x & 31;
  const int hi = lane >> 4, lm = lane & 15;
  const int n = (blockIdx.x * 4 + wave) * 16 + lm;
  const int y = n >> 9, x = n & 511;
  const int mb = blockIdx.y * 16;
  const __bf16* arow = Wp + (size_t)(mb + lm) * (9 * CP2);
  v8f acc = vzero8();
  conv_acc<CP2>(arow, P2, y, x, hi, acc);
#pragma unroll
  for (int g = 0; g < 8; ++g) {
    int m = mb + hi * 8 + g;
    if (m < OC2) out[((size_t)y * NE + x) * RD + m] = acc[g] + bias[m];
  }
}

// ---------------- adjacency: logits -> softmax(R) -> +I -> Laplace ----------------

__global__ void k_rel_logits(const float* __restrict__ rel, const float* __restrict__ wm,
                             const float* __restrict__ bm, float* __restrict__ A0) {
  __shared__ float sw[RH][RD];
  __shared__ float sb[RH];
  for (int i = threadIdx.x; i < RH * RD; i += blockDim.x) sw[i / RD][i % RD] = wm[i];
  if (threadIdx.x < RH) sb[threadIdx.x] = bm[threadIdx.x];
  __syncthreads();
  size_t pix = (size_t)blockIdx.x * blockDim.x + threadIdx.x;
  if (pix >= (size_t)NE * NE) return;
  const float* xp = rel + pix * RD;
  float lg[RH];
#pragma unroll
  for (int r = 0; r < RH; ++r) lg[r] = sb[r];
  for (int d = 0; d < RD; ++d) {
    float v = xp[d];
#pragma unroll
    for (int r = 0; r < RH; ++r) lg[r] += v * sw[r][d];
  }
  float mx = lg[0];
#pragma unroll
  for (int r = 1; r < RH; ++r) mx = fmaxf(mx, lg[r]);
  float s = 0.0f;
#pragma unroll
  for (int r = 0; r < RH; ++r) { lg[r] = __expf(lg[r] - mx); s += lg[r]; }
  const float inv = 1.0f / s;
  const int ii = (int)(pix / NE), jj = (int)(pix % NE);
  const float diag = (ii == jj) ? 1.0f : 0.0f;
#pragma unroll
  for (int r = 0; r < RH; ++r) A0[(size_t)r * NE * NE + pix] = lg[r] * inv + diag;
}

__global__ void k_rowsum(const float* __restrict__ A0, float* __restrict__ dis) {
  const int wave = threadIdx.x >> 5, lane = threadIdx.x & 31;
  const int row = blockIdx.x * 8 + wave;
  if (row >= RH * NE) return;
  const float* p = A0 + (size_t)row * NE;
  float s = 0.0f;
  for (int j = lane; j < NE; j += 32) s += p[j];
#pragma unroll
  for (int m = 16; m; m >>= 1) s += __shfl_xor(s, m);
  if (lane == 0) dis[row] = (s > 0.0f) ? rsqrtf(s) : 0.0f;
}

__global__ void k_scaleA(const float* __restrict__ A0, const float* __restrict__ dis,
                         __bf16* __restrict__ Ab, __bf16* __restrict__ Atb) {
  size_t tot = (size_t)RH * NE * NE;
  size_t i = (size_t)blockIdx.x * blockDim.x + threadIdx.x;
  size_t st = (size_t)gridDim.x * blockDim.x;
  for (; i < tot; i += st) {
    size_t r = i / ((size_t)NE * NE);
    size_t q = i % ((size_t)NE * NE);
    int ii = (int)(q / NE), jj = (int)(q % NE);
    float v = dis[r * NE + ii] * A0[i] * dis[r * NE + jj];
    Ab[i] = (__bf16)v;
    Atb[(r * NE + jj) * NE + ii] = (__bf16)v;
  }
}

// ---------------- GCN head GEMMs ----------------

// H[z] = E(512x768) x Wt[z]^T (Wt NT [96][768]); store transposed bf16 HT[z][96][512]
__global__ void k_gemm_h(const __bf16* __restrict__ E, const __bf16* __restrict__ Wt,
                         __bf16* __restrict__ HT) {
  const int lane = threadIdx.x & 31, hi = lane >> 4, lm = lane & 15;
  const int z = blockIdx.z, mb = blockIdx.y * 16;
  const __bf16* arow = E + (size_t)(mb + lm) * DD;
  const __bf16* brow = Wt + (size_t)z * DR * DD + (size_t)lm * DD;
  v8f acc[6];
#pragma unroll
  for (int t = 0; t < 6; ++t) acc[t] = vzero8();
  reg_gemm_nt<6>(arow, brow, DD, DD, hi, acc);
  __bf16* O = HT + (size_t)z * DR * NE;
#pragma unroll
  for (int t = 0; t < 6; ++t)
#pragma unroll
    for (int g = 0; g < 8; ++g) {
      int m = mb + hi * 8 + g, o = t * 16 + lm;
      O[(size_t)o * NE + m] = (__bf16)acc[t][g];
    }
}

// e_bf[m][z*96+o] = relu( A[z]·hf + A[z]^T·hb + mh_b[z][o] )
__global__ void k_gcn(const __bf16* __restrict__ Ab, const __bf16* __restrict__ Atb,
                      const __bf16* __restrict__ HfT, const __bf16* __restrict__ HbT,
                      const float* __restrict__ bias, __bf16* __restrict__ Eo) {
  const int lane = threadIdx.x & 31, hi = lane >> 4, lm = lane & 15;
  const int z = blockIdx.z, mb = blockIdx.y * 16;
  const __bf16* a1 = Ab + ((size_t)z * NE + mb + lm) * NE;
  const __bf16* a2 = Atb + ((size_t)z * NE + mb + lm) * NE;
  const __bf16* bf = HfT + (size_t)z * DR * NE + (size_t)lm * NE;
  const __bf16* bb = HbT + (size_t)z * DR * NE + (size_t)lm * NE;
  v8f acc[6];
#pragma unroll
  for (int t = 0; t < 6; ++t) acc[t] = vzero8();
  reg_gemm_nt<6>(a1, bf, NE, NE, hi, acc);
  reg_gemm_nt<6>(a2, bb, NE, NE, hi, acc);
#pragma unroll
  for (int t = 0; t < 6; ++t)
#pragma unroll
    for (int g = 0; g < 8; ++g) {
      int m = mb + hi * 8 + g, o = t * 16 + lm;
      float v = fmaxf(acc[t][g] + bias[z * DR + o], 0.0f);
      Eo[(size_t)m * DD + z * DR + o] = (__bf16)v;
    }
}

// ents[n][:] = LN(e2[n][:] + ents[n][:]); also refresh bf16 copy
__global__ void k_add_ln_D(float* __restrict__ ents, const float* __restrict__ e2,
                           const float* __restrict__ g, const float* __restrict__ b,
                           __bf16* __restrict__ Ebf) {
  __shared__ float red[256], red2[256];
  const int n = blockIdx.x, t = threadIdx.x;
  const float* xp = e2 + (size_t)n * DD;
  float* rp = ents + (size_t)n * DD;
  float v[3];
  float s = 0.0f, s2 = 0.0f;
#pragma unroll
  for (int k = 0; k < 3; ++k) {
    int d = t + 256 * k;
    float x = xp[d] + rp[d];
    v[k] = x; s += x; s2 += x * x;
  }
  red[t] = s; red2[t] = s2;
  __syncthreads();
  for (int st = 128; st; st >>= 1) {
    if (t < st) { red[t] += red[t + st]; red2[t] += red2[t + st]; }
    __syncthreads();
  }
  const float mean = red[0] / (float)DD;
  const float var = red2[0] / (float)DD - mean * mean;
  const float rs = rsqrtf(var + EPSL);
#pragma unroll
  for (int k = 0; k < 3; ++k) {
    int d = t + 256 * k;
    float y = (v[k] - mean) * rs * g[d] + b[d];
    rp[d] = y;
    Ebf[(size_t)n * DD + d] = (__bf16)y;
  }
}

// rels_out[i][j][r] = dot(rel[i][j][:], wm[r]) + bm[r]
__global__ void k_final_rels(const float* __restrict__ rel, const float* __restrict__ wm,
                             const float* __restrict__ bm, float* __restrict__ out) {
  __shared__ float sw[RH][RD];
  __shared__ float sb[RH];
  for (int i = threadIdx.x; i < RH * RD; i += blockDim.x) sw[i / RD][i % RD] = wm[i];
  if (threadIdx.x < RH) sb[threadIdx.x] = bm[threadIdx.x];
  __syncthreads();
  size_t pix = (size_t)blockIdx.x * blockDim.x + threadIdx.x;
  if (pix >= (size_t)NE * NE) return;
  const float* xp = rel + pix * RD;
  float lg[RH];
#pragma unroll
  for (int r = 0; r < RH; ++r) lg[r] = sb[r];
  for (int d = 0; d < RD; ++d) {
    float v = xp[d];
#pragma unroll
    for (int r = 0; r < RH; ++r) lg[r] += v * sw[r][d];
  }
#pragma unroll
  for (int r = 0; r < RH; ++r) out[pix * RH + r] = lg[r];
}

// ---------------- host orchestration ----------------

extern "C" void kernel_launch(void* const* d_in, const int* in_sizes, int n_in,
                              void* d_out, int out_size, void* d_ws, size_t ws_size,
                              hipStream_t stream) {
  (void)in_sizes; (void)n_in; (void)out_size; (void)ws_size;
  const float* in_ents = (const float*)d_in[0];
  const float* in_Rloc = (const float*)d_in[1];
  const float* in_wmlp = (const float*)d_in[2];
  const float* in_bmlp = (const float*)d_in[3];
  const float* in_c1w  = (const float*)d_in[4];
  const float* in_c1b  = (const float*)d_in[5];
  const float* in_c2w  = (const float*)d_in[6];
  const float* in_c2b  = (const float*)d_in[7];
  const float* in_pg   = (const float*)d_in[8];
  const float* in_pb   = (const float*)d_in[9];
  const float* in_rg   = (const float*)d_in[10];
  const float* in_rb   = (const float*)d_in[11];
  const float* in_wf   = (const float*)d_in[12];
  const float* in_wb   = (const float*)d_in[13];
  const float* in_mhb  = (const float*)d_in[14];
  const float* in_mw   = (const float*)d_in[15];
  const float* in_mb2  = (const float*)d_in[16];
  const float* in_eg   = (const float*)d_in[17];
  const float* in_eb   = (const float*)d_in[18];

  char* p = (char*)d_ws;
  auto alloc = [&](size_t bytes) -> void* {
    void* r = (void*)p;
    p += (bytes + 255) & ~(size_t)255;
    return r;
  };
  const size_t NN = (size_t)NE * NE;
  const int RCH = 20;  // r-chunk for ReRescal
  float*  bufA  = (float*)alloc(NN * RD * 4);           // relu rel (scratch)
  float*  bufB  = (float*)alloc(NN * RD * 4);           // final rel_embs / prev
  float*  c2out = (float*)alloc(NN * RD * 4);
  __bf16* P1    = (__bf16*)alloc((size_t)NP * NP * CP1 * 2);
  __bf16* P2    = (__bf16*)alloc((size_t)NP * NP * CP2 * 2);
  __bf16* Rt    = (__bf16*)alloc((size_t)RCH * DD * DD * 2);
  __bf16* Tb    = (__bf16*)alloc((size_t)RCH * NE * DD * 2);
  __bf16* Wp1   = (__bf16*)alloc((size_t)OC1P * 9 * CP1 * 2);
  __bf16* Wp2   = (__bf16*)alloc((size_t)OC2P * 9 * CP2 * 2);
  float*  A0    = (float*)alloc((size_t)RH * NN * 4);
  float*  dis   = (float*)alloc((size_t)RH * NE * 4);
  __bf16* Ab    = (__bf16*)alloc((size_t)RH * NN * 2);
  __bf16* Atb   = (__bf16*)alloc((size_t)RH * NN * 2);
  __bf16* WfT   = (__bf16*)alloc((size_t)RH * DR * DD * 2);
  __bf16* WbT   = (__bf16*)alloc((size_t)RH * DR * DD * 2);
  __bf16* HfT   = (__bf16*)alloc((size_t)RH * DR * NE * 2);
  __bf16* HbT   = (__bf16*)alloc((size_t)RH * DR * NE * 2);
  __bf16* Ebf   = (__bf16*)alloc((size_t)NE * DD * 2);
  __bf16* e_bf  = (__bf16*)alloc((size_t)NE * DD * 2);
  __bf16* Wm    = (__bf16*)alloc((size_t)DD * DD * 2);
  float*  e2    = (float*)alloc((size_t)NE * DD * 4);
  float*  ents  = (float*)alloc((size_t)NE * DD * 4);

  (void)hipMemcpyAsync(ents, in_ents, (size_t)NE * DD * 4,
                       hipMemcpyDeviceToDevice, stream);
  k_cvt_bf16<<<256, 256, 0, stream>>>(ents, Ebf, (size_t)NE * DD);
  k_fill0_bf16<<<2048, 256, 0, stream>>>(P2, (size_t)NP * NP * CP2);

  for (int l = 0; l < NL; ++l) {
    // --- ReRescal: rel = relu(E R_r E^T), chunked over r ---
    const float* Rl = in_Rloc + (size_t)l * RD * DD * DD;
    for (int c0 = 0; c0 < RD; c0 += RCH) {
      k_transpose_cvt<<<dim3(DD / 32, DD / 32, RCH), dim3(32, 8, 1), 0, stream>>>(
          Rl + (size_t)c0 * DD * DD, Rt, DD, DD);
      // T = E x Rt^T (bf16 out)
      k_gemm_coop<0><<<dim3(DD / TN, NE / TM, RCH), 128, 0, stream>>>(
          Ebf, Rt, Tb, nullptr, DD, DD, DD, DD,
          0L, (long)DD * DD, (long)NE * DD, 0);
      // rel = relu(T x E^T), scattered to [i][j][r]
      k_gemm_coop<1><<<dim3(NE / TN, NE / TM, RCH), 128, 0, stream>>>(
          Tb, Ebf, bufA, nullptr, DD, DD, DD, 0,
          (long)NE * DD, 0L, 0L, c0);
    }
    if (l > 0)
      k_add_ln_rdim<<<(unsigned)(NN / 8), 256, 0, stream>>>(
          bufA, bufA, bufB, in_rg + (size_t)l * RD, in_rb + (size_t)l * RD);
    // --- posFFN conv (implicit GEMM per tap) ---
    k_pack_w<<<128, 256, 0, stream>>>(in_c1w + (size_t)l * OC1 * RD * 9, Wp1,
                                      OC1P, CP1, OC1, RD);
    k_pack_w<<<128, 256, 0, stream>>>(in_c2w + (size_t)l * OC2 * OC1 * 9, Wp2,
                                      OC2P, CP2, OC2, OC1);
    k_pack_in<<<4096, 256, 0, stream>>>(bufA, P1);
    k_conv1<<<dim3((unsigned)(NN / 64), OC1P / 16, 1), 128, 0, stream>>>(
        Wp1, P1, in_c1b + (size_t)l * OC1, P2);
    k_conv2<<<dim3((unsigned)(NN / 64), OC2P / 16, 1), 128, 0, stream>>>(
        Wp2, P2, in_c2b + (size_t)l * OC2, c2out);
    k_add_ln_rdim<<<(unsigned)(NN / 8), 256, 0, stream>>>(
        bufB, c2out, bufA, in_pg + (size_t)l * RD, in_pb + (size_t)l * RD);
    // --- adjacency ---
    k_rel_logits<<<(unsigned)(NN / 256), 256, 0, stream>>>(bufB, in_wmlp, in_bmlp, A0);
    k_rowsum<<<RH * NE / 8, 256, 0, stream>>>(A0, dis);
    k_scaleA<<<2048, 256, 0, stream>>>(A0, dis, Ab, Atb);
    // --- GCN heads ---
    k_transpose_cvt<<<dim3(DR / 32, DD / 32, RH), dim3(32, 8, 1), 0, stream>>>(
        in_wf + (size_t)l * RH * DD * DR, WfT, DD, DR);
    k_transpose_cvt<<<dim3(DR / 32, DD / 32, RH), dim3(32, 8, 1), 0, stream>>>(
        in_wb + (size_t)l * RH * DD * DR, WbT, DD, DR);
    k_gemm_h<<<dim3(1, NE / 16, RH), 32, 0, stream>>>(Ebf, WfT, HfT);
    k_gemm_h<<<dim3(1, NE / 16, RH), 32, 0, stream>>>(Ebf, WbT, HbT);
    k_gcn<<<dim3(1, NE / 16, RH), 32, 0, stream>>>(Ab, Atb, HfT, HbT,
                                                   in_mhb + (size_t)l * RH * DR, e_bf);
    // --- ent MLP + residual LN ---
    k_cvt_bf16<<<256, 256, 0, stream>>>(in_mw + (size_t)l * DD * DD, Wm,
                                        (size_t)DD * DD);
    k_gemm_coop<2><<<dim3(DD / TN, NE / TM, 1), 128, 0, stream>>>(
        e_bf, Wm, e2, in_mb2 + (size_t)l * DD, DD, DD, DD, DD, 0L, 0L, 0L, 0);
    k_add_ln_D<<<NE, 256, 0, stream>>>(ents, e2, in_eg + (size_t)l * DD,
                                       in_eb + (size_t)l * DD, Ebf);
  }

  (void)hipMemcpyAsync(d_out, ents, (size_t)NE * DD * 4,
                       hipMemcpyDeviceToDevice, stream);
  k_final_rels<<<(unsigned)(NN / 256), 256, 0, stream>>>(
      bufB, in_wmlp, in_bmlp, (float*)d_out + (size_t)NE * DD);
}
